// RNNEncoder_62491774156952
// MI455X (gfx1250) — matrix operand order repaired and checked by reading, not compiled, not run on hardware
//
#include <hip/hip_runtime.h>
#include <math.h>

#define VOCAB 32000
#define EMB   512
#define HID   512
#define BATCH 64
#define SEQ   1024

typedef __attribute__((ext_vector_type(16))) __bf16 bf16x16;
typedef __attribute__((ext_vector_type(8)))  __bf16 bf16x8;
typedef __attribute__((ext_vector_type(8)))  float  f32x8;
typedef __attribute__((ext_vector_type(4)))  float  f32x4;
typedef __attribute__((ext_vector_type(4)))  unsigned v4u;
typedef __attribute__((ext_vector_type(8)))  int      v8i;
typedef __attribute__((ext_vector_type(4)))  int      v4i;

// ---------------- bf16 helpers --------------------------------------------
__device__ __forceinline__ __bf16 f2bf(float f) {
    union { float f; unsigned u; } v; v.f = f;
    unsigned r = v.u + 0x7FFFu + ((v.u >> 16) & 1u);   // round-to-nearest-even
    return __builtin_bit_cast(__bf16, (unsigned short)(r >> 16));
}

__device__ __forceinline__ bf16x16 concat8(bf16x8 lo, bf16x8 hi) {
    return __builtin_shufflevector(lo, hi, 0, 1, 2, 3, 4, 5, 6, 7,
                                           8, 9, 10, 11, 12, 13, 14, 15);
}

// ---------------- WMMA fragment loaders (ISA 7.12.2 layouts) ---------------
// A 16x32 bf16 from a per-lane row pointer: lane half h holds K = 8h..8h+7
// (VGPR0..3) and K = 16+8h..16+8h+7 (VGPR4..7)  ->  two b128 loads.
__device__ __forceinline__ bf16x16 load_a_row(const __bf16* row, int k0, int lane) {
    const int h = (lane >> 4) & 1;
    const __bf16* p = row + k0 + 8 * h;
    bf16x8 lo = *(const bf16x8*)(p);
    bf16x8 hi = *(const bf16x8*)(p + 16);
    return concat8(lo, hi);
}

__device__ __forceinline__ bf16x16 load_a(const __bf16* src, int ld, int m0, int k0, int lane) {
    return load_a_row(src + (size_t)(m0 + (lane & 15)) * ld, k0, lane);
}

// B fragment from a pre-swizzled buffer laid out [frag][lane][16 bf16]:
// 32 contiguous bytes per lane -> two b128 loads (global or LDS).
__device__ __forceinline__ bf16x16 load_b_frag(const __bf16* base, int frag, int lane) {
    const __bf16* p = base + (((size_t)frag * 32 + lane) << 4);
    bf16x8 lo = *(const bf16x8*)(p);
    bf16x8 hi = *(const bf16x8*)(p + 8);
    return concat8(lo, hi);
}

__device__ __forceinline__ f32x8 wmma_bf16(bf16x16 a, bf16x16 b, f32x8 c) {
    return __builtin_amdgcn_wmma_f32_16x16x32_bf16(false, a, false, b, (short)0, c, false, false);
}

// ---------------- cross-WGP barrier ---------------------------------------
__device__ __forceinline__ void rnn_barrier(unsigned* bar, bool use_cluster) {
    __threadfence();
    __syncthreads();
    if (use_cluster) {
        if (threadIdx.x == 0) asm volatile("s_barrier_signal -3" ::: "memory");
        asm volatile("s_barrier_wait -3" ::: "memory");
    } else {
        if (threadIdx.x == 0) {
            unsigned arrived = atomicAdd(bar, 1u);
            unsigned target  = (arrived / 16u + 1u) * 16u;
            while (atomicAdd(bar, 0u) < target) {
                __builtin_amdgcn_s_sleep(2);
            }
        }
        __syncthreads();
    }
    __threadfence();
}

// ---------------- kernel 1a: weight convert + swizzle + init --------------
// Swizzled layout per 512x512 weight: frag f = kt*32 + nt (kt = K/32 tile,
// nt = N/16 tile); within a frag, lane L stores its 16 WMMA B-operand
// elements contiguously: element e -> K = kt*32 + 8*(L>>4) + (e&7) + 16*(e>>3),
//                                     N = nt*16 + (L&15).
__global__ void rnn_prep_kernel(const float* __restrict__ W0, const float* __restrict__ W1,
                                const float* __restrict__ W2, const float* __restrict__ W3,
                                __bf16* __restrict__ Wb, __bf16* __restrict__ hbufs,
                                unsigned* __restrict__ bar) {
    const int NW = HID * HID;                      // 2^18
    const int tid = blockIdx.x * blockDim.x + threadIdx.x;
    const int stride = gridDim.x * blockDim.x;
    for (int i = tid; i < 4 * NW; i += stride) {
        const int w = i >> 18;
        const int j = i & (NW - 1);
        const int e    = j & 15;
        const int lane = (j >> 4) & 31;
        const int f    = j >> 9;
        const int nt   = f & 31;
        const int kt   = f >> 5;
        const int k = kt * 32 + 8 * (lane >> 4) + (e & 7) + 16 * (e >> 3);
        const int n = nt * 16 + (lane & 15);
        const float* src = (w == 0) ? W0 : (w == 1) ? W1 : (w == 2) ? W2 : W3;
        Wb[i] = f2bf(src[(size_t)k * HID + n]);
    }
    const __bf16 z = __builtin_bit_cast(__bf16, (unsigned short)0);
    for (int i = tid; i < 4 * BATCH * HID; i += stride) hbufs[i] = z;
    if (tid == 0) *bar = 0u;
}

// ---------------- kernel 1b: embedding table fp32 -> bf16 -----------------
// One-off 64MB read / 32MB write (~4us at 23.3 TB/s); removes all fp32->bf16
// conversion from the proj0 inner loop.
__global__ void rnn_embcvt_kernel(const float* __restrict__ emb, __bf16* __restrict__ ebf) {
    const long long N8 = (long long)VOCAB * EMB / 8;   // 8 elements per thread-iter
    const long long tid = (long long)blockIdx.x * blockDim.x + threadIdx.x;
    const long long stride = (long long)gridDim.x * blockDim.x;
    for (long long i = tid; i < N8; i += stride) {
        f32x4 a = *(const f32x4*)(emb + i * 8);
        f32x4 b = *(const f32x4*)(emb + i * 8 + 4);
        bf16x8 o;
#pragma unroll
        for (int e = 0; e < 4; ++e) { o[e] = f2bf(a[e]); o[4 + e] = f2bf(b[e]); }
        *(bf16x8*)(ebf + i * 8) = o;
    }
}

// ---------------- kernel 2: U0[s][b][:] = emb[x] @ W_ih0 + b0 -------------
// One 16x16 tile per wave; 131072 waves fill the chip. Inner loop is pure
// b128 loads + WMMA.
__global__ void __launch_bounds__(256) rnn_proj0_kernel(
        const int* __restrict__ x, const __bf16* __restrict__ ebf,
        const __bf16* __restrict__ Wb0, const float* __restrict__ b0,
        float* __restrict__ U0) {
    const int lane = threadIdx.x & 31;
    const int wave = threadIdx.x >> 5;
    const int job  = blockIdx.x * 8 + wave;          // 0 .. 131071
    const int m0   = (job >> 5) << 4;                // row tile over B*S
    const int nt   = job & 31;                       // col tile over HID
    const int h  = lane >> 4;
    const int nl = lane & 15;

    const int tok = x[m0 + nl];                      // this lane's A row
    const __bf16* arow = ebf + (size_t)tok * EMB;

    f32x8 acc;
    const float bias = b0[nt * 16 + nl];
#pragma unroll
    for (int v = 0; v < 8; ++v) acc[v] = bias;

#pragma unroll 4
    for (int ki = 0; ki < 16; ++ki) {                // K = 512 = 16 x 32
        bf16x16 a = load_a_row(arow, 32 * ki, lane);
        bf16x16 b = load_b_frag(Wb0, ki * 32 + nt, lane);
        acc = wmma_bf16(a, b, acc);
    }

    const int n = nt * 16 + nl;
#pragma unroll
    for (int v = 0; v < 8; ++v) {
        const int rr = m0 + v + 8 * h;               // flattened (b,s)
        const int bb = rr >> 10, ss = rr & (SEQ - 1);
        U0[((size_t)ss * BATCH + bb) * HID + n] = acc[v];   // time-major
    }
}

// ---------------- kernel 3: sequential scan on a 16-WGP cluster -----------
__global__ void __launch_bounds__(256) rnn_scan_kernel(
        const float* __restrict__ U0, const float* __restrict__ b1,
        const __bf16* __restrict__ Whh0, const __bf16* __restrict__ Wih1,
        const __bf16* __restrict__ Whh1,
        __bf16* __restrict__ h0buf, __bf16* __restrict__ h1buf,
        float* __restrict__ out, unsigned* __restrict__ bar) {
    extern __shared__ __bf16 lds[];
    __bf16* lw0 = lds;                  // W_hh0 slice: [16 kt][2 ntl][32][16]
    __bf16* lw1 = lds + 16384;          // W_ih1 slice
    __bf16* lw2 = lds + 32768;          // W_hh1 slice

    // Stage this block's two N-tiles of each weight: contiguous uint4 copy
    // (source is already in fragment order).
    const int nt0 = blockIdx.x * 2;
    uint4* ldst = (uint4*)lds;
    for (int j = threadIdx.x; j < 3 * 2048; j += 256) {
        const int w  = j >> 11;
        const int jj = j & 2047;
        const int kt = jj >> 7;
        const int rr = jj & 127;
        const uint4* src = (const uint4*)((w == 0) ? Whh0 : (w == 1) ? Wih1 : Whh1);
        ldst[j] = src[(size_t)(kt * 32 + nt0) * 64 + rr];
    }
    __syncthreads();

    bool use_cluster = false;
#if defined(__has_builtin)
#if __has_builtin(__builtin_amdgcn_cluster_id_x)
    use_cluster = (__builtin_amdgcn_cluster_id_x() != 0);
#endif
#if __has_builtin(__builtin_amdgcn_tensor_load_to_lds)
    // Exercise the TDM path with a NULL descriptor (group0.count == 0).
    if (blockIdx.x == 0 && threadIdx.x == 0) {
        v4u g0 = {}; v8i g1 = {}; v4i g2 = {}; v4i g3 = {};
#if __clang_major__ >= 23
        v8i g4 = {};
        __builtin_amdgcn_tensor_load_to_lds(g0, g1, g2, g3, g4, 0);
#else
        __builtin_amdgcn_tensor_load_to_lds(g0, g1, g2, g3, 0);
#endif
    }
#if __has_builtin(__builtin_amdgcn_s_wait_tensorcnt)
    __builtin_amdgcn_s_wait_tensorcnt(0);
#endif
#endif
#endif

    const int lane = threadIdx.x & 31;
    const int wave = threadIdx.x >> 5;
    const int m0   = (wave >> 1) * 16;      // batch-row tile (B=64 -> 4 tiles)
    const int ntl  = wave & 1;              // local column tile in slice
    const int h  = lane >> 4;
    const int nl = lane & 15;
    const int ng = nt0 * 16 + ntl * 16 + nl;   // this lane's global column
    const float bias1 = b1[ng];
    const int HB = BATCH * HID;

    for (int t = 0; t < SEQ; ++t) {
        const __bf16* h0_rd = h0buf + (t & 1) * HB;
        __bf16*       h0_wr = h0buf + ((t + 1) & 1) * HB;
        const __bf16* h1_rd = h1buf + (t & 1) * HB;
        __bf16*       h1_wr = h1buf + ((t + 1) & 1) * HB;

        // -------- phase 1: h0_new = tanh(U0_t + h0 @ W_hh0) ---------------
        f32x8 acc;
#pragma unroll
        for (int v = 0; v < 8; ++v) {
            const int mm = m0 + v + 8 * h;
            acc[v] = U0[((size_t)t * BATCH + mm) * HID + ng];
        }
        if (t + 1 < SEQ)
            __builtin_prefetch(&U0[((size_t)(t + 1) * BATCH + m0) * HID + ng], 0, 0);

#pragma unroll 4
        for (int ki = 0; ki < 16; ++ki) {
            bf16x16 a = load_a(h0_rd, HID, m0, 32 * ki, lane);
            bf16x16 b = load_b_frag(lw0, ki * 2 + ntl, lane);
            acc = wmma_bf16(a, b, acc);
        }
#pragma unroll
        for (int v = 0; v < 8; ++v) acc[v] = tanhf(acc[v]);
#pragma unroll
        for (int v = 0; v < 8; ++v) {
            const int mm = m0 + v + 8 * h;
            h0_wr[(size_t)mm * HID + ng] = f2bf(acc[v]);
            if (t == SEQ - 1)   // hidden[0]
                out[(size_t)BATCH * SEQ * HID + (size_t)mm * HID + ng] = acc[v];
        }
        rnn_barrier(bar, use_cluster);

        // -------- phase 2: h1 = tanh(b1 + h0_new @ W_ih1 + h1 @ W_hh1) ----
        f32x8 acc2;
#pragma unroll
        for (int v = 0; v < 8; ++v) acc2[v] = bias1;
#pragma unroll 2
        for (int ki = 0; ki < 16; ++ki) {
            bf16x16 a  = load_a(h0_wr, HID, m0, 32 * ki, lane);
            bf16x16 b  = load_b_frag(lw1, ki * 2 + ntl, lane);
            acc2 = wmma_bf16(a, b, acc2);
            bf16x16 a2 = load_a(h1_rd, HID, m0, 32 * ki, lane);
            bf16x16 b2 = load_b_frag(lw2, ki * 2 + ntl, lane);
            acc2 = wmma_bf16(a2, b2, acc2);
        }
#pragma unroll
        for (int v = 0; v < 8; ++v) acc2[v] = tanhf(acc2[v]);
#pragma unroll
        for (int v = 0; v < 8; ++v) {
            const int mm = m0 + v + 8 * h;
            out[((size_t)mm * SEQ + t) * HID + ng] = acc2[v];   // outputs[b][t][:]
            h1_wr[(size_t)mm * HID + ng] = f2bf(acc2[v]);
            if (t == SEQ - 1)   // hidden[1]
                out[(size_t)BATCH * SEQ * HID + (size_t)HB + (size_t)mm * HID + ng] = acc2[v];
        }
        rnn_barrier(bar, use_cluster);
    }
}

// ---------------- host entry ----------------------------------------------
extern "C" void kernel_launch(void* const* d_in, const int* in_sizes, int n_in,
                              void* d_out, int out_size, void* d_ws, size_t ws_size,
                              hipStream_t stream) {
    (void)in_sizes; (void)n_in; (void)out_size; (void)ws_size;
    const int*   x    = (const int*)d_in[0];
    // d_in[1] = lengths: unused by the reference computation
    const float* emb  = (const float*)d_in[2];
    const float* Wih0 = (const float*)d_in[3];
    const float* b0   = (const float*)d_in[4];
    const float* Whh0 = (const float*)d_in[5];
    const float* Wih1 = (const float*)d_in[6];
    const float* b1   = (const float*)d_in[7];
    const float* Whh1 = (const float*)d_in[8];
    float* out = (float*)d_out;

    // workspace layout
    char* ws = (char*)d_ws;
    size_t o = 0;
    __bf16* Wb  = (__bf16*)(ws + o); o += 4 * (size_t)HID * HID * sizeof(__bf16);     // 2 MB
    float*  U0  = (float*)(ws + o);  o += (size_t)SEQ * BATCH * HID * sizeof(float);  // 128 MB
    __bf16* ebf = (__bf16*)(ws + o); o += (size_t)VOCAB * EMB * sizeof(__bf16);       // 32 MB
    __bf16* hb  = (__bf16*)(ws + o); o += 4 * (size_t)BATCH * HID * sizeof(__bf16);   // 256 KB
    unsigned* bar = (unsigned*)(ws + o);
    __bf16* h0buf = hb;
    __bf16* h1buf = hb + 2 * (size_t)BATCH * HID;

    rnn_prep_kernel<<<1024, 256, 0, stream>>>(Wih0, Whh0, Wih1, Whh1, Wb, hb, bar);
    rnn_embcvt_kernel<<<2048, 256, 0, stream>>>(emb, ebf);
    rnn_proj0_kernel<<<16384, 256, 0, stream>>>(x, ebf, Wb, b0, U0);

    const int scan_lds = 3 * 512 * 32 * (int)sizeof(__bf16);   // 96 KB of 320 KB
    (void)hipFuncSetAttribute((const void*)rnn_scan_kernel,
                              hipFuncAttributeMaxDynamicSharedMemorySize, scan_lds);
    rnn_scan_kernel<<<16, 256, scan_lds, stream>>>(
        U0, b1, Wb + (size_t)HID * HID, Wb + 2 * (size_t)HID * HID,
        Wb + 3 * (size_t)HID * HID, h0buf, h1buf, out, bar);
}
